// ContrastiveLoss_6871947673829
// MI455X (gfx1250) — matrix operand order repaired and checked by reading, not compiled, not run on hardware
//
#include <hip/hip_runtime.h>

// ---------------------------------------------------------------------------
// Supervised contrastive loss, fused WMMA pipeline for gfx1250 (MI455X).
//   1) normalize_f16   : L2-normalize rows, pre-scale by sqrt(log2e/0.07),
//                        store f16 copy in workspace (2 MB, L2-resident)
//   2) simloss_wmma    : fused f·f^T (v_wmma_f32_16x16x32_f16), software-
//                        pipelined B tiles (ping-pong), 2 row tiles per wave.
//                        WMMA output is already exp2-scaled -> v_med3 + v_exp_f32.
//                        Diagonal handled analytically (= e^10, subtracted later).
//   3) row_loss_reduce : per-row -log(pos/neg), block partial sums
//   4) final_mean      : deterministic final reduction + NaN guard
// ---------------------------------------------------------------------------

typedef __attribute__((ext_vector_type(16))) _Float16 v16h;
typedef __attribute__((ext_vector_type(8)))  _Float16 v8h;
typedef __attribute__((ext_vector_type(4)))  _Float16 h4;
typedef __attribute__((ext_vector_type(8)))  float    v8f;

#define B_ROWS 8192
#define DIM    128
#define NCHUNK 16
#define COLS_PER_CHUNK  (B_ROWS / NCHUNK)       // 512
#define TILES_PER_CHUNK (COLS_PER_CHUNK / 16)   // 32 (even -> clean ping-pong)
// exp(clip(d/0.07,-10,10)) == exp2(clamp(d*K2, -10*log2e, 10*log2e)),
// K2 = log2(e)/0.07. We bake sqrt(K2) into BOTH operands of the GEMM, so the
// WMMA result is d*K2 directly (no per-element multiply in the hot loop).
#define SQRT_K2 4.5398168f       // sqrt(1.4426950408889634 / 0.07)
#define L2CLP   14.4269504f      // 10 * log2(e)
#define EXP10   22026.4657948f   // e^10 : exact diagonal contribution

// ---- 1) L2-normalize each row (x sqrt(K2)), emit f16 -----------------------
__global__ __launch_bounds__(256) void normalize_f16(const float* __restrict__ f,
                                                     _Float16* __restrict__ fn) {
  const int row  = blockIdx.x * 8 + (threadIdx.x >> 5);
  const int lane = threadIdx.x & 31;
  const float4 x = ((const float4*)(f + (size_t)row * DIM))[lane];
  float ss = x.x * x.x + x.y * x.y + x.z * x.z + x.w * x.w;
#pragma unroll
  for (int m = 16; m >= 1; m >>= 1) ss += __shfl_xor(ss, m, 32);
  const float inv = SQRT_K2 / fmaxf(sqrtf(ss), 1e-8f);
  h4 o;
  o[0] = (_Float16)(x.x * inv);
  o[1] = (_Float16)(x.y * inv);
  o[2] = (_Float16)(x.z * inv);
  o[3] = (_Float16)(x.w * inv);
  *(h4*)(fn + (size_t)row * DIM + lane * 4) = o;
}

// ---- helpers for the fused kernel ------------------------------------------
__device__ __forceinline__ void loadA(const _Float16* __restrict__ fn, int rowBase,
                                      int lo, int hi, v16h a[4]) {
  // A layout (matches B): lane half 0 holds K {0..15} of each 32-chunk,
  // lane half 1 holds K {16..31}.
  const _Float16* arow = fn + (size_t)(rowBase + lo) * DIM;
#pragma unroll
  for (int kk = 0; kk < 4; ++kk) {
    v8h alo = *(const v8h*)(arow + kk * 32 + hi * 16);
    v8h ahi = *(const v8h*)(arow + kk * 32 + hi * 16 + 8);
    a[kk] = __builtin_shufflevector(alo, ahi, 0, 1, 2, 3, 4, 5, 6, 7,
                                    8, 9, 10, 11, 12, 13, 14, 15);
  }
}

__device__ __forceinline__ void loadB(const _Float16* __restrict__ fn, int colBase,
                                      int lo, int hi, v16h b[4]) {
  // B = f^T (col-major of f): lane's column = colBase+lo, contiguous K per half.
  const _Float16* brow = fn + (size_t)(colBase + lo) * DIM + hi * 16;
#pragma unroll
  for (int kk = 0; kk < 4; ++kk) b[kk] = *(const v16h*)(brow + kk * 32);
}

__device__ __forceinline__ void processTile(const v16h a0[4], const v16h a1[4],
                                            const v16h b[4], int clab,
                                            const int rlab0[8], const int rlab1[8],
                                            float psum0[8], float nsum0[8],
                                            float psum1[8], float nsum1[8]) {
  v8f c0 = {}, c1 = {};
#pragma unroll
  for (int kk = 0; kk < 4; ++kk) {
    c0 = __builtin_amdgcn_wmma_f32_16x16x32_f16(false, a0[kk], false, b[kk],
                                                (short)0, c0, false, false);
    c1 = __builtin_amdgcn_wmma_f32_16x16x32_f16(false, a1[kk], false, b[kk],
                                                (short)0, c1, false, false);
  }
  // c already equals sim * (log2e/0.07). Inputs are finite -> raw v_med3_num
  // clamp (no NaN-canonicalizing self-max) feeding v_exp_f32.
#pragma unroll
  for (int r = 0; r < 8; ++r) {
    const float e0 = __builtin_amdgcn_exp2f(__builtin_amdgcn_fmed3f(c0[r], -L2CLP, L2CLP));
    const float e1 = __builtin_amdgcn_exp2f(__builtin_amdgcn_fmed3f(c1[r], -L2CLP, L2CLP));
    nsum0[r] += e0;
    nsum1[r] += e1;
    psum0[r] += (clab == rlab0[r]) ? e0 : 0.0f;
    psum1[r] += (clab == rlab1[r]) ? e1 : 0.0f;
  }
}

// ---- 2) fused sim + exp + masked row sums ----------------------------------
// One wave per (32-row strip, column chunk). B tiles ping-pong double-buffered.
__global__ __launch_bounds__(32) void simloss_wmma(const _Float16* __restrict__ fn,
                                                   const int* __restrict__ labels,
                                                   float* __restrict__ posPart,
                                                   float* __restrict__ negPart) {
  const int lane = threadIdx.x;
  const int lo = lane & 15;
  const int hi = lane >> 4;
  const int rowBase0 = blockIdx.x * 32;
  const int rowBase1 = rowBase0 + 16;
  const int chunk    = blockIdx.y;

  v16h a0[4], a1[4];
  loadA(fn, rowBase0, lo, hi, a0);
  loadA(fn, rowBase1, lo, hi, a1);

  int rlab0[8], rlab1[8];
#pragma unroll
  for (int r = 0; r < 8; ++r) {
    rlab0[r] = labels[rowBase0 + hi * 8 + r];   // C/D: VGPR r -> row r + 8*half
    rlab1[r] = labels[rowBase1 + hi * 8 + r];
  }

  float psum0[8], nsum0[8], psum1[8], nsum1[8];
#pragma unroll
  for (int r = 0; r < 8; ++r) {
    psum0[r] = 0.0f; nsum0[r] = 0.0f; psum1[r] = 0.0f; nsum1[r] = 0.0f;
  }

  const int colStart = chunk * COLS_PER_CHUNK;

  v16h bufA[4], bufB[4];
  int clabA, clabB;
  loadB(fn, colStart, lo, hi, bufA);
  clabA = labels[colStart + lo];

  for (int t = 0; t < TILES_PER_CHUNK; t += 2) {
    // issue loads for tile t+1 before consuming tile t
    const int cb1 = colStart + (t + 1) * 16;
    loadB(fn, cb1, lo, hi, bufB);
    clabB = labels[cb1 + lo];
    processTile(a0, a1, bufA, clabA, rlab0, rlab1, psum0, nsum0, psum1, nsum1);

    // issue loads for tile t+2 (dummy in-range reload on the last iteration)
    const int cb2 = (t + 2 < TILES_PER_CHUNK) ? colStart + (t + 2) * 16 : colStart;
    loadB(fn, cb2, lo, hi, bufA);
    clabA = labels[cb2 + lo];
    processTile(a0, a1, bufB, clabB, rlab0, rlab1, psum0, nsum0, psum1, nsum1);
  }

  // Reduce across the 16 lanes of each half (xor masks < 16 stay in-half).
#pragma unroll
  for (int r = 0; r < 8; ++r) {
#pragma unroll
    for (int m = 8; m >= 1; m >>= 1) {
      psum0[r] += __shfl_xor(psum0[r], m, 32);
      nsum0[r] += __shfl_xor(nsum0[r], m, 32);
      psum1[r] += __shfl_xor(psum1[r], m, 32);
      nsum1[r] += __shfl_xor(nsum1[r], m, 32);
    }
  }
  if (lo == 0) {
#pragma unroll
    for (int r = 0; r < 8; ++r) {
      const size_t base = (size_t)chunk * B_ROWS;
      posPart[base + rowBase0 + hi * 8 + r] = psum0[r];
      negPart[base + rowBase0 + hi * 8 + r] = nsum0[r];
      posPart[base + rowBase1 + hi * 8 + r] = psum1[r];
      negPart[base + rowBase1 + hi * 8 + r] = nsum1[r];
    }
  }
}

// ---- 3) per-row loss + per-block partial sum (deterministic) ---------------
__global__ __launch_bounds__(256) void row_loss_reduce(const float* __restrict__ posPart,
                                                       const float* __restrict__ negPart,
                                                       float* __restrict__ blockSums) {
  const int row = blockIdx.x * 256 + threadIdx.x;
  float p = 0.0f, n = 0.0f;
#pragma unroll
  for (int c = 0; c < NCHUNK; ++c) {
    p += posPart[(size_t)c * B_ROWS + row];
    n += negPart[(size_t)c * B_ROWS + row];
  }
  // remove the analytically-known diagonal term (clipped self-sim -> e^10)
  p = fmaxf(p - EXP10, 1e-8f);
  n = fmaxf(n - EXP10, 1e-8f);
  float loss = __logf(n) - __logf(p);   // -log(p/n)
#pragma unroll
  for (int m = 16; m >= 1; m >>= 1) loss += __shfl_xor(loss, m, 32);

  __shared__ float wsum[8];
  if ((threadIdx.x & 31) == 0) wsum[threadIdx.x >> 5] = loss;
  __syncthreads();
  if (threadIdx.x == 0) {
    float s = 0.0f;
#pragma unroll
    for (int w = 0; w < 8; ++w) s += wsum[w];
    blockSums[blockIdx.x] = s;
  }
}

// ---- 4) final mean + NaN guard ---------------------------------------------
__global__ __launch_bounds__(32) void final_mean(const float* __restrict__ blockSums,
                                                 float* __restrict__ out) {
  float v = blockSums[threadIdx.x];   // exactly 32 block sums
#pragma unroll
  for (int m = 16; m >= 1; m >>= 1) v += __shfl_xor(v, m, 32);
  if (threadIdx.x == 0) {
    const float mean = v * (1.0f / (float)B_ROWS);
    out[0] = (mean != mean) ? 0.0f : mean;
  }
}

// ---------------------------------------------------------------------------
extern "C" void kernel_launch(void* const* d_in, const int* in_sizes, int n_in,
                              void* d_out, int out_size, void* d_ws, size_t ws_size,
                              hipStream_t stream) {
  (void)in_sizes; (void)n_in; (void)out_size; (void)ws_size;
  const float* features = (const float*)d_in[0];
  const int*   labels   = (const int*)d_in[1];
  float*       out      = (float*)d_out;

  char* ws = (char*)d_ws;
  _Float16* fn     = (_Float16*)ws;                                 // 2 MB
  float* posPart   = (float*)(ws + (size_t)B_ROWS * DIM * 2);       // 512 KB
  float* negPart   = posPart + (size_t)NCHUNK * B_ROWS;             // 512 KB
  float* blockSums = negPart + (size_t)NCHUNK * B_ROWS;             // 128 B

  normalize_f16<<<B_ROWS / 8, 256, 0, stream>>>(features, fn);
  simloss_wmma<<<dim3(B_ROWS / 32, NCHUNK), 32, 0, stream>>>(fn, labels,
                                                             posPart, negPart);
  row_loss_reduce<<<B_ROWS / 256, 256, 0, stream>>>(posPart, negPart, blockSums);
  final_mean<<<1, 32, 0, stream>>>(blockSums, out);
}
